// CrossAttention_25795573580081
// MI455X (gfx1250) — matrix operand order repaired and checked by reading, not compiled, not run on hardware
//
#include <hip/hip_runtime.h>
#include <hip/hip_bf16.h>

typedef __attribute__((ext_vector_type(16))) _Float16 v16h;
typedef __attribute__((ext_vector_type(8)))  float    v8f;
typedef __attribute__((ext_vector_type(4)))  unsigned int u32x4;
typedef __attribute__((ext_vector_type(4)))  int      i32x4;

#if defined(__has_builtin)
#if __has_builtin(__builtin_amdgcn_global_load_async_to_lds_b128)
#define USE_ASYNC_LDS 1
#endif
#endif

namespace {
constexpr int   NH      = 8;
constexpr int   HD      = 96;
constexpr int   AHS     = 768;
constexpr int   SKV     = 4165;
constexpr int   SKV_PAD = 4192;
constexpr int   NB      = 2;
constexpr float QK_SCALE = 0.10206207261596575f; // 1/sqrt(96)

union Frag { v16h h; u32x4 u[2]; };

__device__ inline v16h ld2(const _Float16* p0, const _Float16* p1) {
  Frag f;
  f.u[0] = *reinterpret_cast<const u32x4*>(p0);
  f.u[1] = *reinterpret_cast<const u32x4*>(p1);
  return f.h;
}

__device__ inline v8f wmma_f16(v16h a, v16h b, v8f c) {
  // (neg_a, A, neg_b, B, c_mod, C, reuse_a, reuse_b)
  return __builtin_amdgcn_wmma_f32_16x16x32_f16(false, a, false, b, (short)0, c,
                                                false, false);
}

__device__ inline v8f vzero8() {
  v8f z = {0.f, 0.f, 0.f, 0.f, 0.f, 0.f, 0.f, 0.f};
  return z;
}

// 16-byte global -> LDS copy (gfx1250 async path when available)
__device__ inline void cp16_g2l(_Float16* l, const _Float16* g) {
#ifdef USE_ASYNC_LDS
  __builtin_amdgcn_global_load_async_to_lds_b128(
      (__attribute__((address_space(1))) i32x4*)(_Float16*)g,
      (__attribute__((address_space(3))) i32x4*)l, 0, 0);
#else
  *reinterpret_cast<u32x4*>(l) = *reinterpret_cast<const u32x4*>(g);
#endif
}

__device__ inline void wait_async_copies() {
#ifdef USE_ASYNC_LDS
#if __has_builtin(__builtin_amdgcn_s_wait_asynccnt)
  __builtin_amdgcn_s_wait_asynccnt(0);
#else
  asm volatile("s_wait_asynccnt 0x0" ::: "memory");
#endif
#endif
}
} // namespace

// ---------------- conversion kernels ----------------

__global__ void conv_f32_to_f16(const float* __restrict__ src,
                                _Float16* __restrict__ dst, int n) {
  int i = blockIdx.x * blockDim.x + threadIdx.x;
  if (i < n) dst[i] = (_Float16)src[i];
}

// src [K,N] row-major f32 -> dst [N,K] row-major f16 (i.e. transposed)
__global__ void transpose_to_f16(const float* __restrict__ src,
                                 _Float16* __restrict__ dst, int K, int N) {
  int i = blockIdx.x * blockDim.x + threadIdx.x;
  if (i < K * N) {
    int k = i / N, n = i % N;
    dst[(size_t)n * K + k] = (_Float16)src[i];
  }
}

// ---------------- generic WMMA GEMM ----------------
// C[M,N] = A[M,K] * BT[N,K]^T + bias[n]
// mode 0: Q  -> out16 [B,H,SKV,96], value * scale (QK scale folded into Q)
// mode 1: K  -> out16 [B,H,SKV_PAD,96]
// mode 2: V  -> out16 [B,H,96,SKV_PAD] (transposed store)
// mode 3: O  -> out32 [M,N] fp32 (final output segment)
__global__ void __launch_bounds__(128)
gemm_kernel(const _Float16* __restrict__ A, const _Float16* __restrict__ BT,
            const float* __restrict__ bias, _Float16* __restrict__ out16,
            float* __restrict__ out32, int M, int N, int Kd,
            long a_bs, long a_rs, int seq, int kvoff, float scale, int mode) {
  const int lane  = threadIdx.x & 31;
  const int wave  = threadIdx.x >> 5;
  const int mtile = (blockIdx.x * 4 + wave) * 16;
  const int ntile = blockIdx.y * 64;
  if (mtile >= M) return;                       // wave-uniform
  const int l16 = lane & 15, lh = lane >> 4;

  int ar = mtile + l16; if (ar > M - 1) ar = M - 1;
  const _Float16* Arow = A + (size_t)(ar / seq) * a_bs + (size_t)(ar % seq) * a_rs;

  v8f acc[4] = {vzero8(), vzero8(), vzero8(), vzero8()};

  for (int kc = 0; kc < Kd; kc += 32) {
    const _Float16* ap = Arow + kc + 8 * lh;     // A-layout: lane<16 K 0-7/16-23
    const v16h af = ld2(ap, ap + 16);
#pragma unroll
    for (int j = 0; j < 4; ++j) {
      int n = ntile + j * 16 + l16; if (n > N - 1) n = N - 1;
      const _Float16* bp = BT + (size_t)n * Kd + kc + 8 * lh; // B: lane=col
      acc[j] = wmma_f16(af, ld2(bp, bp + 16), acc[j]);
    }
  }

#pragma unroll
  for (int j = 0; j < 4; ++j) {
#pragma unroll
    for (int r = 0; r < 8; ++r) {
      const int row = mtile + r + 8 * lh;        // C-layout row
      const int n   = ntile + j * 16 + l16;      // C-layout col
      if (row < M && n < N) {
        const float v = acc[j][r] + bias[n];
        if (mode == 3) {
          out32[(size_t)row * N + n] = v;
        } else {
          const int b = row / seq, t = row % seq;
          const int hh = n / HD, dc = n % HD;
          if (mode == 2)
            out16[((size_t)(b * NH + hh) * HD + dc) * SKV_PAD + kvoff + t] =
                (_Float16)v;
          else if (mode == 1)
            out16[((size_t)(b * NH + hh) * SKV_PAD + kvoff + t) * HD + dc] =
                (_Float16)v;
          else
            out16[((size_t)(b * NH + hh) * SKV + kvoff + t) * HD + dc] =
                (_Float16)(v * scale);
        }
      }
    }
  }
}

// ---------------- flash attention ----------------
// Block = 4 waves = 64 consecutive q rows of one (b,h).
// Each 32-key step: K/V tiles staged once into LDS (async copy), shared by
// all 4 waves; per-wave online softmax; P routed C->A layout via LDS.
__global__ void __launch_bounds__(128)
attn_kernel(const _Float16* __restrict__ Q, const _Float16* __restrict__ Kb,
            const _Float16* __restrict__ VT, _Float16* __restrict__ ctx) {
  __shared__ _Float16 ldsK[32 * HD];       // [key][d]   6KB
  __shared__ _Float16 ldsV[HD * 32];       // [d][key]   6KB
  __shared__ _Float16 ldsP[4][16 * 32];    // per-wave P 4KB
  const int tid   = threadIdx.x;
  const int lane  = tid & 31;
  const int wave  = tid >> 5;
  const int nqb   = (SKV + 63) / 64;       // 66 blocks of 64 q-rows per (b,h)
  const int bh    = blockIdx.x / nqb;
  const int qt0   = (blockIdx.x % nqb) * 64 + wave * 16;
  const int l16 = lane & 15, lh = lane >> 4;

  const _Float16* Qp = Q  + (size_t)bh * SKV * HD;
  const _Float16* Kp = Kb + (size_t)bh * SKV_PAD * HD;
  const _Float16* Vp = VT + (size_t)bh * HD * SKV_PAD;

  int qr = qt0 + l16; if (qr > SKV - 1) qr = SKV - 1;
  v16h qa[3];
#pragma unroll
  for (int c = 0; c < 3; ++c) {
    const _Float16* ap = Qp + (size_t)qr * HD + c * 32 + 8 * lh;
    qa[c] = ld2(ap, ap + 16);
  }

  v8f acc[6];
#pragma unroll
  for (int t6 = 0; t6 < 6; ++t6) acc[t6] = vzero8();
  float mrun[8], lrun[8];
#pragma unroll
  for (int r = 0; r < 8; ++r) { mrun[r] = -1e30f; lrun[r] = 0.f; }

  _Float16* P = &ldsP[wave][0];

  for (int kb = 0; kb < SKV; kb += 32) {
    // ---- stage K (flat 6KB: rows kb..kb+31 contiguous) and V tile ----
    const _Float16* kg = Kp + (size_t)kb * HD;
#pragma unroll
    for (int j = 0; j < 3; ++j) {
      const int c = tid + 128 * j;                 // 384 16B chunks each
      cp16_g2l(ldsK + c * 8, kg + c * 8);
      const int dr = c >> 2, part = c & 3;         // V^T rows strided
      cp16_g2l(ldsV + dr * 32 + part * 8,
               Vp + (size_t)dr * SKV_PAD + kb + part * 8);
    }
    wait_async_copies();
    __syncthreads();

    const bool v0 = (kb + l16)      < SKV;
    const bool v1 = (kb + 16 + l16) < SKV;

    v8f s0 = vzero8(), s1 = vzero8();
#pragma unroll
    for (int c = 0; c < 3; ++c) {                  // D=96 in 3 k-chunks
      const _Float16* b0 = ldsK + (size_t)l16 * HD        + c * 32 + 8 * lh;
      const _Float16* b1 = ldsK + (size_t)(16 + l16) * HD + c * 32 + 8 * lh;
      s0 = wmma_f16(qa[c], ld2(b0, b0 + 16), s0);
      s1 = wmma_f16(qa[c], ld2(b1, b1 + 16), s1);
    }

    float corr[8], mnew[8];
#pragma unroll
    for (int r = 0; r < 8; ++r) {
      float a = v0 ? s0[r] : -1e30f;
      float b = v1 ? s1[r] : -1e30f;
      float t = fmaxf(a, b);
#pragma unroll
      for (int m = 1; m < 16; m <<= 1) t = fmaxf(t, __shfl_xor(t, m, 32));
      mnew[r] = fmaxf(mrun[r], t);
      corr[r] = __expf(mrun[r] - mnew[r]);
      mrun[r] = mnew[r];
    }

#pragma unroll
    for (int r = 0; r < 8; ++r) {
      float p0 = v0 ? __expf(s0[r] - mnew[r]) : 0.f;
      float p1 = v1 ? __expf(s1[r] - mnew[r]) : 0.f;
      float t = p0 + p1;
#pragma unroll
      for (int m = 1; m < 16; m <<= 1) t += __shfl_xor(t, m, 32);
      lrun[r] = lrun[r] * corr[r] + t;
      // C-layout -> row-major 16x32 tile in per-wave LDS region
      P[(r + 8 * lh) * 32 + l16]      = (_Float16)p0;
      P[(r + 8 * lh) * 32 + 16 + l16] = (_Float16)p1;
    }

#pragma unroll
    for (int t6 = 0; t6 < 6; ++t6) {
#pragma unroll
      for (int r = 0; r < 8; ++r) acc[t6][r] *= corr[r];
    }

    // reload P as A-fragment (same-wave LDS ops are in-order)
    const _Float16* pp = P + l16 * 32 + 8 * lh;
    const v16h pa = ld2(pp, pp + 16);

#pragma unroll
    for (int t6 = 0; t6 < 6; ++t6) {               // PV from staged V tile
      const _Float16* vp = ldsV + (size_t)(t6 * 16 + l16) * 32 + 8 * lh;
      acc[t6] = wmma_f16(pa, ld2(vp, vp + 16), acc[t6]);
    }
    __syncthreads();                                // protect staged tiles
  }

  const int b = bh / NH, hh = bh % NH;
#pragma unroll
  for (int t6 = 0; t6 < 6; ++t6) {
#pragma unroll
    for (int r = 0; r < 8; ++r) {
      int row = qt0 + r + 8 * lh;
      if (row < SKV) {
        float v = acc[t6][r] / lrun[r];
        ctx[((size_t)b * SKV + row) * AHS + hh * HD + t6 * 16 + l16] = (_Float16)v;
      }
    }
  }
}

// ---------------- host launcher ----------------

extern "C" void kernel_launch(void* const* d_in, const int* in_sizes, int n_in,
                              void* d_out, int out_size, void* d_ws, size_t ws_size,
                              hipStream_t stream) {
  (void)in_sizes; (void)n_in; (void)out_size; (void)ws_size;
  static const int  DIMS[4]   = {96, 192, 384, 768};
  static const int  SEQS[4]   = {3136, 784, 196, 49};
  static const int  QOFF[4]   = {0, 3136, 3920, 4116};
  static const long OUTOFF[4] = {0, 602112, 903168, 1053696};

  char* ws = (char*)d_ws;
  size_t off = 0;
  auto take = [&](size_t bytes) -> char* {
    char* p = ws + off;
    off = (off + bytes + 255) & ~(size_t)255;
    return p;
  };

  _Float16* h16[4]; _Float16* wqT[4]; _Float16* wkT[4];
  _Float16* wvT[4]; _Float16* woT[4];
  for (int i = 0; i < 4; ++i) h16[i] = (_Float16*)take((size_t)2 * SEQS[i] * DIMS[i] * 2);
  for (int i = 0; i < 4; ++i) wqT[i] = (_Float16*)take((size_t)AHS * DIMS[i] * 2);
  for (int i = 0; i < 4; ++i) wkT[i] = (_Float16*)take((size_t)AHS * DIMS[i] * 2);
  for (int i = 0; i < 4; ++i) wvT[i] = (_Float16*)take((size_t)AHS * DIMS[i] * 2);
  for (int i = 0; i < 4; ++i) woT[i] = (_Float16*)take((size_t)DIMS[i] * AHS * 2);
  _Float16* Qb  = (_Float16*)take((size_t)NB * NH * SKV * HD * 2);
  _Float16* Kbf = (_Float16*)take((size_t)NB * NH * SKV_PAD * HD * 2);
  _Float16* VTb = (_Float16*)take((size_t)NB * NH * HD * SKV_PAD * 2);
  _Float16* ctx = (_Float16*)take((size_t)NB * SKV * AHS * 2);

  for (int i = 0; i < 4; ++i) {
    const float* h  = (const float*)d_in[i * 9 + 0];
    const float* Wq = (const float*)d_in[i * 9 + 1];
    const float* bq = (const float*)d_in[i * 9 + 2];
    const float* Wk = (const float*)d_in[i * 9 + 3];
    const float* bk = (const float*)d_in[i * 9 + 4];
    const float* Wv = (const float*)d_in[i * 9 + 5];
    const float* bv = (const float*)d_in[i * 9 + 6];
    const float* Wo = (const float*)d_in[i * 9 + 7];
    const int d = DIMS[i], s = SEQS[i];

    const int nh = 2 * s * d;
    conv_f32_to_f16<<<(nh + 255) / 256, 256, 0, stream>>>(h, h16[i], nh);
    const int nw = d * AHS;
    transpose_to_f16<<<(nw + 255) / 256, 256, 0, stream>>>(Wq, wqT[i], d, AHS);
    transpose_to_f16<<<(nw + 255) / 256, 256, 0, stream>>>(Wk, wkT[i], d, AHS);
    transpose_to_f16<<<(nw + 255) / 256, 256, 0, stream>>>(Wv, wvT[i], d, AHS);
    transpose_to_f16<<<(nw + 255) / 256, 256, 0, stream>>>(Wo, woT[i], AHS, d);

    const int M = 2 * s;
    dim3 g((M + 63) / 64, AHS / 64);
    gemm_kernel<<<g, 128, 0, stream>>>(h16[i], wqT[i], bq, Qb, nullptr, M, AHS, d,
                                       (long)s * d, d, s, QOFF[i], QK_SCALE, 0);
    gemm_kernel<<<g, 128, 0, stream>>>(h16[i], wkT[i], bk, Kbf, nullptr, M, AHS, d,
                                       (long)s * d, d, s, QOFF[i], 1.0f, 1);
    gemm_kernel<<<g, 128, 0, stream>>>(h16[i], wvT[i], bv, VTb, nullptr, M, AHS, d,
                                       (long)s * d, d, s, QOFF[i], 1.0f, 2);
  }

  {
    const int nqb  = (SKV + 63) / 64;                 // 66
    const int nblk = NB * NH * nqb;                   // 1056
    attn_kernel<<<nblk, 128, 0, stream>>>(Qb, Kbf, VTb, ctx);
  }

  for (int i = 0; i < 4; ++i) {
    const float* bo = (const float*)d_in[i * 9 + 8];
    const int d = DIMS[i], s = SEQS[i], M = 2 * s;
    dim3 g((M + 63) / 64, (d + 63) / 64);
    gemm_kernel<<<g, 128, 0, stream>>>(ctx + (size_t)QOFF[i] * AHS, woT[i], bo,
                                       nullptr, (float*)d_out + OUTOFF[i],
                                       M, d, AHS, (long)SKV * AHS, AHS, s, 0,
                                       1.0f, 3);
  }
}
